// DIEN_32049045963136
// MI455X (gfx1250) — matrix operand order repaired and checked by reading, not compiled, not run on hardware
//
#include <hip/hip_runtime.h>
#include <hip/hip_bf16.h>
#include <math.h>

typedef __attribute__((ext_vector_type(16))) _Float16 v16h;
typedef __attribute__((ext_vector_type(8)))  _Float16 v8h;
typedef __attribute__((ext_vector_type(8)))  float    v8f;
typedef _Float16 h16;

#define B_   1024
#define T_   200
#define E_   32
#define D_   64
#define H_   64
#define TH3  192           // 3*H
#define MAUX (B_*(T_-1))   // 203776

__device__ __forceinline__ float sigmf(float x) { return 1.0f / (1.0f + __expf(-x)); }

// ---------------- WMMA fragment helpers (layouts per CDNA5 ISA 7.12.2) ----------------
// A 16x32 f16: lanes 0-15 -> rows, elems 0-7 = K(k0..k0+7), elems 8-15 = K(k0+16..k0+23);
//              lanes 16-31 -> same rows, +8 on both K runs. Two contiguous 16B loads/lane.
__device__ __forceinline__ v16h ld_a(const h16* A, int lda, int lane, int k0) {
  int row = lane & 15; int kh = (lane >> 4) * 8;
  const h16* p = A + (size_t)row * lda + k0 + kh;
  v8h lo = *(const v8h*)p;
  v8h hi = *(const v8h*)(p + 16);
  v16h a;
#pragma unroll
  for (int i = 0; i < 8; ++i) { a[i] = lo[i]; a[i + 8] = hi[i]; }
  return a;
}
// B 32x16 f16 from TRANSPOSED storage Bt[N][K]: lane = column, same K layout as A:
// elems 0-7 = K(k0+8h..), elems 8-15 = K(k0+16+8h..) -> two contiguous 16B loads per lane.
__device__ __forceinline__ v16h ld_bt(const h16* Bt, int ldk, int lane, int k0) {
  int col = lane & 15; int kb = k0 + ((lane >> 4) << 3);
  const h16* p = Bt + (size_t)col * ldk + kb;
  v8h lo = *(const v8h*)p;
  v8h hi = *(const v8h*)(p + 16);
  v16h b;
#pragma unroll
  for (int i = 0; i < 8; ++i) { b[i] = lo[i]; b[i + 8] = hi[i]; }
  return b;
}
__device__ __forceinline__ v8f wmma_f16(v16h a, v16h b, v8f c) {
  return __builtin_amdgcn_wmma_f32_16x16x32_f16(false, a, false, b, (short)0, c, false, false);
}

// ---------------- generic WMMA GEMM: 64x16 output per wave ----------------
// C[M,N] = act(A[M,K]_f16 @ B + bias); B given transposed as Bt[N][K].
// grid = (M/64, N/16), block = 32 (one wave, 4 row tiles share each B fragment).
__global__ void wmma_gemm_kernel(const h16* __restrict__ A, int lda,
                                 const h16* __restrict__ Bt, int K,
                                 float* __restrict__ Cf, int ldc,
                                 h16* __restrict__ Ch, int ldch,
                                 const float* __restrict__ bias, int act) {
  int lane = threadIdx.x & 31;
  size_t row0 = (size_t)blockIdx.x * 64;
  int    col0 = blockIdx.y * 16;
  const h16* Bp = Bt + (size_t)col0 * K;
  v8f acc[4] = {{}, {}, {}, {}};
  for (int k = 0; k < K; k += 32) {
    if (k + 32 < K) __builtin_prefetch((const void*)(A + row0 * (size_t)lda + k + 32), 0, 1);
    v16h b = ld_bt(Bp, K, lane, k);
#pragma unroll
    for (int mt = 0; mt < 4; ++mt) {
      v16h a = ld_a(A + (row0 + mt * 16) * (size_t)lda, lda, lane, k);
      acc[mt] = wmma_f16(a, b, acc[mt]);
    }
  }
  int col = lane & 15; int rb = (lane >> 4) * 8;
  float bv = bias ? bias[col0 + col] : 0.0f;
#pragma unroll
  for (int mt = 0; mt < 4; ++mt) {
#pragma unroll
    for (int r = 0; r < 8; ++r) {
      float v = acc[mt][r] + bv;
      if (act == 1) v = sigmf(v);
      else if (act == 2) v = fmaxf(v, 0.0f);
      size_t row = row0 + mt * 16 + rb + r;
      if (Cf) Cf[row * (size_t)ldc + col0 + col] = v;
      if (Ch) Ch[row * (size_t)ldch + col0 + col] = (h16)v;
    }
  }
}

// ---------------- persistent GRU / AUGRU scan ----------------
// grid = B/16 blocks, 128 threads (4 waves). Each block owns 16 batch rows;
// h kept in LDS (f16), Wh (transposed [192][64]) held as 6 B fragments/wave in regs.
__global__ void rnn_scan_kernel(const float* __restrict__ xp,   // [B*T, 192] x@Wx+b
                                const h16*  __restrict__ Wht,   // [192][64] f16 transposed
                                const float* __restrict__ att,  // [B,T] or null (null => GRU)
                                float* __restrict__ out_seq,    // [B,T,64] or null
                                h16*  __restrict__ out_seq_h,   // [B,T,64] or null
                                float* __restrict__ h_final) {  // [B,64] or null
  __shared__ h16 hs[16][72];
  int lane = threadIdx.x & 31;
  int w    = threadIdx.x >> 5;      // 0..3 -> 16-col slice of H
  int b0   = blockIdx.x * 16;
  int col  = lane & 15;
  int rb   = (lane >> 4) * 8;
  int jcol = w * 16 + col;

  v16h bz[2], br[2], bn[2];
#pragma unroll
  for (int s = 0; s < 2; ++s) {
    bz[s] = ld_bt(Wht + (size_t)((w + 0) * 16) * H_, H_, lane, s * 32);
    br[s] = ld_bt(Wht + (size_t)((w + 4) * 16) * H_, H_, lane, s * 32);
    bn[s] = ld_bt(Wht + (size_t)((w + 8) * 16) * H_, H_, lane, s * 32);
  }
  for (int i = threadIdx.x; i < 16 * 72; i += blockDim.x) (&hs[0][0])[i] = (h16)0.0f;
  __syncthreads();

  for (int t = 0; t < T_; ++t) {
    v16h a0 = ld_a(&hs[0][0], 72, lane, 0);
    v16h a1 = ld_a(&hs[0][0], 72, lane, 32);
    v8f az = {}, ar = {}, an = {};
    az = wmma_f16(a0, bz[0], az); az = wmma_f16(a1, bz[1], az);
    ar = wmma_f16(a0, br[0], ar); ar = wmma_f16(a1, br[1], ar);
    an = wmma_f16(a0, bn[0], an); an = wmma_f16(a1, bn[1], an);

    float hn[8];
#pragma unroll
    for (int r = 0; r < 8; ++r) {
      int row = rb + r; int gb = b0 + row;
      size_t xo = ((size_t)gb * T_ + t) * TH3;
      float hprev = (float)hs[row][jcol];
      float zz = sigmf(xp[xo + jcol]       + az[r]);
      float rr = sigmf(xp[xo + 64 + jcol]  + ar[r]);
      float nn = tanhf(xp[xo + 128 + jcol] + rr * an[r]);
      float hv;
      if (att) { float u = att[(size_t)gb * T_ + t] * zz; hv = (1.0f - u) * hprev + u * nn; }
      else     { hv = zz * hprev + (1.0f - zz) * nn; }
      hn[r] = hv;
      if (out_seq)   out_seq  [((size_t)gb * T_ + t) * H_ + jcol] = hv;
      if (out_seq_h) out_seq_h[((size_t)gb * T_ + t) * H_ + jcol] = (h16)hv;
    }
    __syncthreads();
#pragma unroll
    for (int r = 0; r < 8; ++r) hs[rb + r][jcol] = (h16)hn[r];
    __syncthreads();
  }
  if (h_final) {
#pragma unroll
    for (int r = 0; r < 8; ++r)
      h_final[(size_t)(b0 + rb + r) * H_ + jcol] = (float)hs[rb + r][jcol];
  }
}

// ---------------- small helper kernels ----------------
// Convert f32 weight [K,N] (row-major) to padded TRANSPOSED f16 [dN][dK].
__global__ void cvt_pad_f16_t(const float* __restrict__ src, h16* __restrict__ dst,
                              int K, int N, int dK, int dN) {
  int idx = blockIdx.x * blockDim.x + threadIdx.x;
  if (idx >= dN * dK) return;
  int n = idx / dK, k = idx % dK;
  dst[idx] = (k < K && n < N) ? (h16)src[k * N + n] : (h16)0.0f;
}

__global__ void pad_bias(const float* __restrict__ src, float* __restrict__ dst, int n, int dn) {
  int i = blockIdx.x * blockDim.x + threadIdx.x;
  if (i < dn) dst[i] = (i < n) ? src[i] : 0.0f;
}

__global__ void gather_clicks(const int* __restrict__ item_ids, const int* __restrict__ cate_ids,
                              const float* __restrict__ item_emb, const float* __restrict__ cate_emb,
                              h16* __restrict__ beh_h) {
  size_t idx = (size_t)blockIdx.x * blockDim.x + threadIdx.x;
  if (idx >= (size_t)B_ * T_ * D_) return;
  int j = (int)(idx & 63); size_t bt = idx >> 6;
  float v = (j < 32) ? item_emb[(size_t)item_ids[bt] * E_ + j]
                     : cate_emb[(size_t)cate_ids[bt] * E_ + (j - 32)];
  beh_h[idx] = (h16)v;
}

__global__ void gather_misc(const int* __restrict__ uid, const int* __restrict__ tiid,
                            const int* __restrict__ tcid,
                            const float* __restrict__ user_emb, const float* __restrict__ item_emb,
                            const float* __restrict__ cate_emb,
                            float* __restrict__ target, float* __restrict__ uprof) {
  int idx = blockIdx.x * blockDim.x + threadIdx.x;
  if (idx < B_ * D_) {
    int b = idx >> 6, j = idx & 63;
    target[idx] = (j < 32) ? item_emb[(size_t)tiid[b] * E_ + j]
                           : cate_emb[(size_t)tcid[b] * E_ + (j - 32)];
  }
  if (idx < B_ * E_) {
    int b = idx >> 5, j = idx & 31;
    uprof[idx] = user_emb[(size_t)uid[b] * E_ + j];
  }
}

__global__ void build_aux_in(const h16* __restrict__ gseq_h, const h16* __restrict__ beh_h,
                             h16* __restrict__ aux_in) {
  size_t idx = (size_t)blockIdx.x * blockDim.x + threadIdx.x;
  if (idx >= (size_t)MAUX * 128) return;
  int j = (int)(idx & 127); size_t m = idx >> 7;
  int b = (int)(m / (T_ - 1)); int tt = (int)(m % (T_ - 1));
  h16 v = (j < 64) ? gseq_h[((size_t)b * T_ + tt) * H_ + j]
                   : beh_h[((size_t)b * T_ + tt + 1) * D_ + (j - 64)];
  aux_in[idx] = v;
}

__global__ void attn_scores(const float* __restrict__ target, const float* __restrict__ gseq,
                            float* __restrict__ scores) {
  int idx = blockIdx.x * blockDim.x + threadIdx.x;
  if (idx >= B_ * T_) return;
  int b = idx / T_;
  const float* tg = target + (size_t)b * D_;
  const float* g  = gseq + (size_t)idx * H_;
  float s = 0.0f;
#pragma unroll
  for (int j = 0; j < H_; ++j) s += tg[j] * g[j];
  scores[idx] = s;
}

__global__ void softmax_T(float* __restrict__ scores) {   // in place, block per batch row
  __shared__ float red[256];
  int b = blockIdx.x, tid = threadIdx.x;
  float v = (tid < T_) ? scores[(size_t)b * T_ + tid] : -1e30f;
  red[tid] = v; __syncthreads();
  for (int s = 128; s > 0; s >>= 1) { if (tid < s) red[tid] = fmaxf(red[tid], red[tid + s]); __syncthreads(); }
  float mx = red[0]; __syncthreads();
  float e = (tid < T_) ? __expf(v - mx) : 0.0f;
  red[tid] = e; __syncthreads();
  for (int s = 128; s > 0; s >>= 1) { if (tid < s) red[tid] += red[tid + s]; __syncthreads(); }
  if (tid < T_) scores[(size_t)b * T_ + tid] = e / red[0];
}

__global__ void aux_tail(const float* __restrict__ a2, const float* __restrict__ w3,
                         const float* __restrict__ b3, float* __restrict__ nlp) {
  size_t m = (size_t)blockIdx.x * blockDim.x + threadIdx.x;
  if (m >= (size_t)MAUX) return;
  const float* row = a2 + m * 64;
  float l0 = b3[0], l1 = b3[1];
  for (int k = 0; k < 50; ++k) { float v = row[k]; l0 += v * w3[2 * k]; l1 += v * w3[2 * k + 1]; }
  float mx = fmaxf(l0, l1);
  float e0 = __expf(l0 - mx), e1 = __expf(l1 - mx);
  nlp[m] = -__logf(e0 / (e0 + e1));
}

__global__ void reduce_mean(const float* __restrict__ v, int n, float* __restrict__ out) {
  __shared__ float red[256];
  float s = 0.0f;
  for (int i = threadIdx.x; i < n; i += 256) s += v[i];
  red[threadIdx.x] = s; __syncthreads();
  for (int st = 128; st > 0; st >>= 1) { if (threadIdx.x < st) red[threadIdx.x] += red[threadIdx.x + st]; __syncthreads(); }
  if (threadIdx.x == 0) out[0] = red[0] / (float)n;
}

__global__ void build_join(const float* __restrict__ h_aug, const float* __restrict__ uprof,
                           float* __restrict__ join) {
  int idx = blockIdx.x * blockDim.x + threadIdx.x;
  if (idx >= B_ * 96) return;
  int b = idx / 96, j = idx % 96;
  join[idx] = (j < 64) ? h_aug[(size_t)b * 64 + j] : uprof[(size_t)b * 32 + (j - 64)];
}

__global__ void bn_stats(const float* __restrict__ join, float* __restrict__ mu, float* __restrict__ rsig) {
  int c = threadIdx.x;
  if (c >= 96) return;
  float s = 0.0f, s2 = 0.0f;
  for (int b = 0; b < B_; ++b) { float v = join[(size_t)b * 96 + c]; s += v; s2 += v * v; }
  float m = s / (float)B_;
  float var = s2 / (float)B_ - m * m;
  mu[c] = m; rsig[c] = rsqrtf(var + 1e-3f);
}

__global__ void bn_apply(const float* __restrict__ join, const float* __restrict__ mu,
                         const float* __restrict__ rsig, const float* __restrict__ gamma,
                         const float* __restrict__ beta, h16* __restrict__ xn) {
  int idx = blockIdx.x * blockDim.x + threadIdx.x;
  if (idx >= B_ * 96) return;
  int c = idx % 96;
  xn[idx] = (h16)((join[idx] - mu[c]) * rsig[c] * gamma[c] + beta[c]);
}

__global__ void head_final(const float* __restrict__ y2, const float* __restrict__ w3,
                           const float* __restrict__ b3, float* __restrict__ out) {
  int b = blockIdx.x * blockDim.x + threadIdx.x;
  if (b >= B_) return;
  float l0 = b3[0], l1 = b3[1];
  for (int k = 0; k < 80; ++k) { float v = y2[(size_t)b * 80 + k]; l0 += v * w3[2 * k]; l1 += v * w3[2 * k + 1]; }
  float mx = fmaxf(l0, l1);
  float e0 = __expf(l0 - mx), e1 = __expf(l1 - mx), s = e0 + e1;
  out[2 * b] = e0 / s; out[2 * b + 1] = e1 / s;           // softmax output
  out[2 * B_ + 2 * b] = l0; out[2 * B_ + 2 * b + 1] = l1; // logits
}

// ---------------- host-side orchestration ----------------
extern "C" void kernel_launch(void* const* d_in, const int* in_sizes, int n_in,
                              void* d_out, int out_size, void* d_ws, size_t ws_size,
                              hipStream_t stream) {
  (void)in_sizes; (void)n_in; (void)out_size; (void)ws_size;
  const int*   user_id   = (const int*)d_in[0];
  const int*   tgt_item  = (const int*)d_in[1];
  const int*   tgt_cate  = (const int*)d_in[2];
  const int*   clk_item  = (const int*)d_in[3];
  const int*   clk_cate  = (const int*)d_in[4];
  const float* user_emb  = (const float*)d_in[5];
  const float* item_emb  = (const float*)d_in[6];
  const float* cate_emb  = (const float*)d_in[7];
  const float* gru_Wx    = (const float*)d_in[8];
  const float* gru_Wh    = (const float*)d_in[9];
  const float* gru_b     = (const float*)d_in[10];
  const float* aug_Wx    = (const float*)d_in[11];
  const float* aug_Wh    = (const float*)d_in[12];
  const float* aug_b     = (const float*)d_in[13];
  const float* aux_w1    = (const float*)d_in[14];
  const float* aux_b1    = (const float*)d_in[15];
  const float* aux_w2    = (const float*)d_in[16];
  const float* aux_b2    = (const float*)d_in[17];
  const float* aux_w3    = (const float*)d_in[18];
  const float* aux_b3    = (const float*)d_in[19];
  const float* bn_gamma  = (const float*)d_in[20];
  const float* bn_beta   = (const float*)d_in[21];
  const float* fc_w1     = (const float*)d_in[22];
  const float* fc_b1     = (const float*)d_in[23];
  const float* fc_w2     = (const float*)d_in[25];
  const float* fc_b2     = (const float*)d_in[26];
  const float* fc_w3     = (const float*)d_in[28];
  const float* fc_b3     = (const float*)d_in[29];
  float* out = (float*)d_out;

  // workspace carve
  char* base = (char*)d_ws; size_t off = 0;
  auto alloc = [&](size_t bytes) -> void* { void* p = base + off; off += (bytes + 255) & ~(size_t)255; return p; };
  h16*   beh_h   = (h16*)  alloc((size_t)B_ * T_ * D_ * 2);
  float* target  = (float*)alloc((size_t)B_ * D_ * 4);
  float* uprof   = (float*)alloc((size_t)B_ * E_ * 4);
  float* xp      = (float*)alloc((size_t)B_ * T_ * TH3 * 4);   // reused: GRU xp, then AUGRU xp
  float* gseq    = (float*)alloc((size_t)B_ * T_ * H_ * 4);
  h16*   gseq_h  = (h16*)  alloc((size_t)B_ * T_ * H_ * 2);
  float* att     = (float*)alloc((size_t)B_ * T_ * 4);
  float* h_aug   = (float*)alloc((size_t)B_ * H_ * 4);
  h16*   aux_in  = (h16*)  alloc((size_t)MAUX * 128 * 2);
  h16*   a1_h    = (h16*)  alloc((size_t)MAUX * 128 * 2);
  float* a2      = (float*)alloc((size_t)MAUX * 64 * 4);
  float* nlp     = (float*)alloc((size_t)MAUX * 4);
  float* join    = (float*)alloc((size_t)B_ * 96 * 4);
  float* mu      = (float*)alloc(96 * 4);
  float* rsig    = (float*)alloc(96 * 4);
  h16*   xn_h    = (h16*)  alloc((size_t)B_ * 96 * 2);
  h16*   y1_h    = (h16*)  alloc((size_t)B_ * 224 * 2);
  float* y2      = (float*)alloc((size_t)B_ * 80 * 4);
  // all weights stored TRANSPOSED f16: [N][K]
  h16*   wgru_x  = (h16*)  alloc(192 * 64 * 2);
  h16*   wgru_h  = (h16*)  alloc(192 * 64 * 2);
  h16*   waug_x  = (h16*)  alloc(192 * 64 * 2);
  h16*   waug_h  = (h16*)  alloc(192 * 64 * 2);
  h16*   waux1   = (h16*)  alloc(128 * 128 * 2);
  h16*   waux2   = (h16*)  alloc(64 * 128 * 2);
  h16*   wfc1    = (h16*)  alloc(224 * 96 * 2);
  h16*   wfc2    = (h16*)  alloc(80 * 224 * 2);
  float* baux1   = (float*)alloc(128 * 4);
  float* baux2   = (float*)alloc(64 * 4);
  float* bfc1    = (float*)alloc(224 * 4);

  auto cdiv = [](size_t a, size_t b) { return (unsigned)((a + b - 1) / b); };

  // weights -> transposed padded f16
  cvt_pad_f16_t<<<cdiv(192 * 64, 256), 256, 0, stream>>>(gru_Wx, wgru_x, 64, 192, 64, 192);
  cvt_pad_f16_t<<<cdiv(192 * 64, 256), 256, 0, stream>>>(gru_Wh, wgru_h, 64, 192, 64, 192);
  cvt_pad_f16_t<<<cdiv(192 * 64, 256), 256, 0, stream>>>(aug_Wx, waug_x, 64, 192, 64, 192);
  cvt_pad_f16_t<<<cdiv(192 * 64, 256), 256, 0, stream>>>(aug_Wh, waug_h, 64, 192, 64, 192);
  cvt_pad_f16_t<<<cdiv(128 * 128, 256), 256, 0, stream>>>(aux_w1, waux1, 128, 100, 128, 128);
  cvt_pad_f16_t<<<cdiv(64 * 128, 256), 256, 0, stream>>>(aux_w2, waux2, 100, 50, 128, 64);
  cvt_pad_f16_t<<<cdiv(224 * 96, 256), 256, 0, stream>>>(fc_w1, wfc1, 96, 200, 96, 224);
  cvt_pad_f16_t<<<cdiv(80 * 224, 256), 256, 0, stream>>>(fc_w2, wfc2, 200, 80, 224, 80);
  pad_bias<<<1, 128, 0, stream>>>(aux_b1, baux1, 100, 128);
  pad_bias<<<1, 64, 0, stream>>>(aux_b2, baux2, 50, 64);
  pad_bias<<<1, 224, 0, stream>>>(fc_b1, bfc1, 200, 224);

  // embedding gathers
  gather_clicks<<<cdiv((size_t)B_ * T_ * D_, 256), 256, 0, stream>>>(clk_item, clk_cate, item_emb, cate_emb, beh_h);
  gather_misc<<<cdiv(B_ * D_, 256), 256, 0, stream>>>(user_id, tgt_item, tgt_cate, user_emb, item_emb, cate_emb, target, uprof);

  // xp_gru = beh @ gru_Wx + b   [B*T,192]
  wmma_gemm_kernel<<<dim3((B_ * T_) / 64, TH3 / 16), 32, 0, stream>>>(
      beh_h, D_, wgru_x, D_, xp, TH3, (h16*)nullptr, 0, gru_b, 0);

  // GRU scan -> gseq / gseq_h
  rnn_scan_kernel<<<B_ / 16, 128, 0, stream>>>(xp, wgru_h, (const float*)nullptr, gseq, gseq_h, (float*)nullptr);

  // attention over GRU outputs
  attn_scores<<<cdiv(B_ * T_, 256), 256, 0, stream>>>(target, gseq, att);
  softmax_T<<<B_, 256, 0, stream>>>(att);

  // aux MLP on [gru_out[:,:-1], beh[:,1:]]
  build_aux_in<<<cdiv((size_t)MAUX * 128, 256), 256, 0, stream>>>(gseq_h, beh_h, aux_in);
  wmma_gemm_kernel<<<dim3(MAUX / 64, 128 / 16), 32, 0, stream>>>(
      aux_in, 128, waux1, 128, (float*)nullptr, 0, a1_h, 128, baux1, 1);
  wmma_gemm_kernel<<<dim3(MAUX / 64, 64 / 16), 32, 0, stream>>>(
      a1_h, 128, waux2, 128, a2, 64, (h16*)nullptr, 0, baux2, 1);
  aux_tail<<<cdiv(MAUX, 256), 256, 0, stream>>>(a2, aux_w3, aux_b3, nlp);
  reduce_mean<<<1, 256, 0, stream>>>(nlp, MAUX, out + 4 * B_);  // aux_loss at d_out[4096]

  // xp_aug = gru_out @ aug_Wx + b (reuse xp buffer)
  wmma_gemm_kernel<<<dim3((B_ * T_) / 64, TH3 / 16), 32, 0, stream>>>(
      gseq_h, H_, waug_x, H_, xp, TH3, (h16*)nullptr, 0, aug_b, 0);

  // AUGRU scan -> h_aug
  rnn_scan_kernel<<<B_ / 16, 128, 0, stream>>>(xp, waug_h, att, (float*)nullptr, (h16*)nullptr, h_aug);

  // head: BN -> fc1(relu) -> fc2(relu) -> fc3 -> softmax
  build_join<<<cdiv(B_ * 96, 256), 256, 0, stream>>>(h_aug, uprof, join);
  bn_stats<<<1, 128, 0, stream>>>(join, mu, rsig);
  bn_apply<<<cdiv(B_ * 96, 256), 256, 0, stream>>>(join, mu, rsig, bn_gamma, bn_beta, xn_h);
  wmma_gemm_kernel<<<dim3(B_ / 64, 224 / 16), 32, 0, stream>>>(
      xn_h, 96, wfc1, 96, (float*)nullptr, 0, y1_h, 224, bfc1, 2);
  wmma_gemm_kernel<<<dim3(B_ / 64, 80 / 16), 32, 0, stream>>>(
      y1_h, 224, wfc2, 224, y2, 80, (h16*)nullptr, 0, fc_b2, 2);
  head_final<<<cdiv(B_, 256), 256, 0, stream>>>(y2, fc_w3, fc_b3, out);
}
